// Megnet_NodeModel_82343112999493
// MI455X (gfx1250) — matrix operand order repaired and checked by reading, not compiled, not run on hardware
//
#include <hip/hip_runtime.h>

typedef _Float16 half16 __attribute__((ext_vector_type(16)));
typedef _Float16 half8  __attribute__((ext_vector_type(8)));
typedef float    float8 __attribute__((ext_vector_type(8)));
typedef float    float4v __attribute__((ext_vector_type(4)));

#define NODE_D 64
#define EDGE_D 48
#define U_D    32
#define IN_D   144      // 64 + 48 + 32
#define KPAD   160      // pad K of layer-1 to 5 tiles of 32
#define HID    128
#define OUT_D  64
#define WAVES  4
#define ROWS_PER_BLK (WAVES * 16)

// ---------------------------------------------------------------- WMMA helpers

__device__ __forceinline__ float8 wmma_f16(half16 a, half16 b, float8 c) {
  // D = A(16x32 f16) * B(32x16 f16) + C(16x16 f32)
  return __builtin_amdgcn_wmma_f32_16x16x32_f16(
      /*neg_a=*/false, a, /*neg_b=*/false, b,
      /*c_mod=*/(short)0, c, /*reuse_a=*/false, /*reuse_b=*/false);
}

// A-fragment (16x32 f16) from an LDS row-major [16][HID] tile.
// lane<16 : elems 0..7 = K[k0..k0+7],    elems 8..15 = K[k0+16..k0+23]
// lane>=16: elems 0..7 = K[k0+8..k0+15], elems 8..15 = K[k0+24..k0+31]
__device__ __forceinline__ half16 load_a_lds(const _Float16* rowbase, int kt, int hi) {
  const _Float16* p = rowbase + kt * 32 + hi * 8;
  half8 lo = *(const half8*)p;
  half8 hh = *(const half8*)(p + 16);
  return __builtin_shufflevector(lo, hh, 0,1,2,3,4,5,6,7,8,9,10,11,12,13,14,15);
}

// B-fragment (32x16 f16) from LDS W^T [N][Kstride]:
// lane<16: col=lane, K[k0..k0+7],K[k0+16..k0+23]; lane>=16: col=lane-16, K[k0+8..k0+15],K[k0+24..k0+31]
__device__ __forceinline__ half16 load_b_lds(const _Float16* wt, int kstride,
                                             int nt, int kt, int lane15, int hi) {
  const _Float16* p = wt + (size_t)(nt * 16 + lane15) * kstride + kt * 32 + hi * 8;
  half8 lo = *(const half8*)p;
  half8 hh = *(const half8*)(p + 16);
  return __builtin_shufflevector(lo, hh, 0,1,2,3,4,5,6,7,8,9,10,11,12,13,14,15);
}

// ---------------------------------------------------------------- kernels

__global__ __launch_bounds__(256) void zero_kernel(float* __restrict__ p, long n) {
  long i = (long)blockIdx.x * blockDim.x + threadIdx.x;
  if (i < n) p[i] = 0.0f;
}

__global__ __launch_bounds__(256) void scatter_kernel(
    const float* __restrict__ edge_attr, const int* __restrict__ edge_src,
    float* __restrict__ sums, float* __restrict__ cnt, int n_edges) {
  int e = blockIdx.x * blockDim.x + threadIdx.x;
  if (e >= n_edges) return;
  int s = edge_src[e];                         // edge_index row 0
  // Pull the whole 192B row in as b128 loads first, then fire the atomics.
  float4v buf[12];
  const float4v* rp = (const float4v*)(edge_attr + (size_t)e * EDGE_D);
#pragma unroll
  for (int i = 0; i < 12; ++i) buf[i] = rp[i];
  unsafeAtomicAdd(&cnt[s], 1.0f);              // -> global_atomic_add_f32 (hits L2)
  float* dst = sums + (size_t)s * EDGE_D;
#pragma unroll
  for (int i = 0; i < 12; ++i)
#pragma unroll
    for (int j = 0; j < 4; ++j) unsafeAtomicAdd(&dst[i * 4 + j], buf[i][j]);
}

__global__ __launch_bounds__(128) void mlp_kernel(
    const float* __restrict__ x, const float* __restrict__ u,
    const int* __restrict__ batch,
    const float* __restrict__ sums, const float* __restrict__ cnt,
    const float* __restrict__ W1, const float* __restrict__ b1,
    const float* __restrict__ W2, const float* __restrict__ b2,
    const float* __restrict__ W3, const float* __restrict__ b3,
    float* __restrict__ out, int n_nodes) {
  __shared__ _Float16 Wlds[HID * KPAD];        // 40 KB: W1^T / W2^T / W3^T (phased)
  __shared__ _Float16 hbuf[WAVES][16 * HID];   // 16 KB: per-wave activation tile

  const int tid    = threadIdx.x;
  const int wave   = tid >> 5;
  const int lane   = tid & 31;
  const int lane15 = lane & 15;
  const int hi     = lane >> 4;                // 0: lanes 0-15, 1: lanes 16-31

  // ---- phase 0: stage W1^T (f32->f16, K padded to 160 with zeros)
  for (int idx = tid; idx < HID * KPAD; idx += 128) {
    int n = idx & (HID - 1);
    int k = idx >> 7;                          // 0..159
    Wlds[(size_t)n * KPAD + k] = (k < IN_D) ? (_Float16)W1[(size_t)k * HID + n]
                                            : (_Float16)0.0f;
  }
  __syncthreads();

  // ---- build layer-1 A fragments straight from global (no LDS round trip)
  long node  = (long)blockIdx.x * ROWS_PER_BLK + wave * 16 + lane15;
  long nodec = node < n_nodes ? node : (long)(n_nodes - 1);
  const float* xr = x + (size_t)nodec * NODE_D;
  const float* sr = sums + (size_t)nodec * EDGE_D;
  float inv = __builtin_amdgcn_rcpf(fmaxf(cnt[nodec], 1.0f));   // v_rcp_f32
  const float* ur = u + (size_t)batch[nodec] * U_D;

  // 8-aligned run of 8 consecutive features; regions are 8-multiples so a run
  // never straddles a region boundary -> two global_load_b128 per run.
  auto feat8 = [&](int k) -> float8 {
    float8 r;
    if (k < NODE_D) {
      float4v a = *(const float4v*)(xr + k);
      float4v b = *(const float4v*)(xr + k + 4);
      r = float8{a[0], a[1], a[2], a[3], b[0], b[1], b[2], b[3]};
    } else if (k < NODE_D + EDGE_D) {
      const float* p = sr + (k - NODE_D);
      float4v a = *(const float4v*)p;
      float4v b = *(const float4v*)(p + 4);
      r = float8{a[0] * inv, a[1] * inv, a[2] * inv, a[3] * inv,
                 b[0] * inv, b[1] * inv, b[2] * inv, b[3] * inv};
    } else if (k < IN_D) {
      const float* p = ur + (k - NODE_D - EDGE_D);
      float4v a = *(const float4v*)p;
      float4v b = *(const float4v*)(p + 4);
      r = float8{a[0], a[1], a[2], a[3], b[0], b[1], b[2], b[3]};
    } else {
      r = (float8)(0.0f);
    }
    return r;
  };

  half16 a1[5];
#pragma unroll
  for (int t = 0; t < 5; ++t) {
    int kb = t * 32 + hi * 8;
    float8 lo = feat8(kb);
    float8 hh = feat8(kb + 16);
#pragma unroll
    for (int i = 0; i < 8; ++i) {
      a1[t][i]     = (_Float16)lo[i];
      a1[t][i + 8] = (_Float16)hh[i];
    }
  }

  // ---- layer 1: [16,160] @ [160,128]
  {
    float8 acc[8];
#pragma unroll
    for (int nt = 0; nt < 8; ++nt) acc[nt] = (float8)(0.0f);
#pragma unroll
    for (int kt = 0; kt < 5; ++kt)
#pragma unroll
      for (int nt = 0; nt < 8; ++nt)
        acc[nt] = wmma_f16(a1[kt], load_b_lds(Wlds, KPAD, nt, kt, lane15, hi), acc[nt]);
#pragma unroll
    for (int nt = 0; nt < 8; ++nt) {
      float bias = b1[nt * 16 + lane15];
#pragma unroll
      for (int r = 0; r < 8; ++r) {
        float v = acc[nt][r] + bias;
        v = v > 0.0f ? v : 0.0f;
        hbuf[wave][(size_t)(r + hi * 8) * HID + nt * 16 + lane15] = (_Float16)v;
      }
    }
  }

  __syncthreads();
  // ---- stage W2^T
  for (int idx = tid; idx < HID * HID; idx += 128) {
    int n = idx & (HID - 1);
    int k = idx >> 7;
    Wlds[(size_t)n * HID + k] = (_Float16)W2[(size_t)k * HID + n];
  }
  __syncthreads();

  const _Float16* arow = &hbuf[wave][(size_t)lane15 * HID];

  // ---- layer 2: [16,128] @ [128,128]
  {
    half16 a2[4];
#pragma unroll
    for (int kt = 0; kt < 4; ++kt) a2[kt] = load_a_lds(arow, kt, hi);

    float8 acc[8];
#pragma unroll
    for (int nt = 0; nt < 8; ++nt) acc[nt] = (float8)(0.0f);
#pragma unroll
    for (int kt = 0; kt < 4; ++kt)
#pragma unroll
      for (int nt = 0; nt < 8; ++nt)
        acc[nt] = wmma_f16(a2[kt], load_b_lds(Wlds, HID, nt, kt, lane15, hi), acc[nt]);
#pragma unroll
    for (int nt = 0; nt < 8; ++nt) {
      float bias = b2[nt * 16 + lane15];
#pragma unroll
      for (int r = 0; r < 8; ++r) {
        float v = acc[nt][r] + bias;
        v = v > 0.0f ? v : 0.0f;
        hbuf[wave][(size_t)(r + hi * 8) * HID + nt * 16 + lane15] = (_Float16)v;
      }
    }
  }

  __syncthreads();
  // ---- stage W3^T (64 output cols)
  for (int idx = tid; idx < OUT_D * HID; idx += 128) {
    int n = idx & (OUT_D - 1);
    int k = idx >> 6;
    Wlds[(size_t)n * HID + k] = (_Float16)W3[(size_t)k * OUT_D + n];
  }
  __syncthreads();

  // ---- layer 3: [16,128] @ [128,64] -> global store (f32)
  {
    half16 a3[4];
#pragma unroll
    for (int kt = 0; kt < 4; ++kt) a3[kt] = load_a_lds(arow, kt, hi);

    float8 acc[4];
#pragma unroll
    for (int nt = 0; nt < 4; ++nt) acc[nt] = (float8)(0.0f);
#pragma unroll
    for (int kt = 0; kt < 4; ++kt)
#pragma unroll
      for (int nt = 0; nt < 4; ++nt)
        acc[nt] = wmma_f16(a3[kt], load_b_lds(Wlds, HID, nt, kt, lane15, hi), acc[nt]);

#pragma unroll
    for (int nt = 0; nt < 4; ++nt) {
      float bias = b3[nt * 16 + lane15];
#pragma unroll
      for (int r = 0; r < 8; ++r) {
        long nodem = (long)blockIdx.x * ROWS_PER_BLK + wave * 16 + r + hi * 8;
        if (nodem < n_nodes)
          out[(size_t)nodem * OUT_D + nt * 16 + lane15] = acc[nt][r] + bias;
      }
    }
  }
}

// ---------------------------------------------------------------- launcher

extern "C" void kernel_launch(void* const* d_in, const int* in_sizes, int n_in,
                              void* d_out, int out_size, void* d_ws, size_t ws_size,
                              hipStream_t stream) {
  const float* x          = (const float*)d_in[0];
  const int*   edge_index = (const int*)d_in[1];   // [2, E]; row 0 = source
  const float* edge_attr  = (const float*)d_in[2];
  const float* u          = (const float*)d_in[3];
  const int*   batch      = (const int*)d_in[4];
  const float* W1 = (const float*)d_in[5];
  const float* b1 = (const float*)d_in[6];
  const float* W2 = (const float*)d_in[7];
  const float* b2 = (const float*)d_in[8];
  const float* W3 = (const float*)d_in[9];
  const float* b3 = (const float*)d_in[10];
  float* out = (float*)d_out;

  const int n_nodes = in_sizes[0] / NODE_D;
  const int n_edges = in_sizes[2] / EDGE_D;

  float* sums = (float*)d_ws;                          // [n_nodes, 48]
  float* cnt  = sums + (size_t)n_nodes * EDGE_D;       // [n_nodes]

  long zn = (long)n_nodes * (EDGE_D + 1);
  zero_kernel<<<(int)((zn + 255) / 256), 256, 0, stream>>>(sums, zn);

  scatter_kernel<<<(n_edges + 255) / 256, 256, 0, stream>>>(
      edge_attr, edge_index, sums, cnt, n_edges);

  int blocks = (n_nodes + ROWS_PER_BLK - 1) / ROWS_PER_BLK;
  mlp_kernel<<<blocks, WAVES * 32, 0, stream>>>(
      x, u, batch, sums, cnt, W1, b1, W2, b2, W3, b3, out, n_nodes);
}
